// GINEncoder_12721693130772
// MI455X (gfx1250) — matrix operand order repaired and checked by reading, not compiled, not run on hardware
//
#include <hip/hip_runtime.h>
#include <hip/hip_bf16.h>
#include <stdint.h>

// Problem constants (match reference)
#define N_NODES 50000
#define N_EDGES 250000
#define DIM     300
#define NLAYERS 5
#define NB      2048
#define NHID    512
#define NOUT    768
#define EPS_F   1e-5f

#define BN_CHUNKS 125
#define BN_ROWS_PER_CHUNK 400   // 125 * 400 = 50000

typedef __attribute__((ext_vector_type(16))) __bf16 v16bf;
typedef __attribute__((ext_vector_type(8)))  __bf16 v8bf;
typedef __attribute__((ext_vector_type(8)))  float  v8f;
typedef __attribute__((ext_vector_type(4)))  unsigned int u32x4;
typedef __attribute__((ext_vector_type(8)))  int i32x8;
typedef __attribute__((ext_vector_type(4)))  int i32x4;

__device__ __forceinline__ unsigned short f2bf(float x) {
  unsigned u = __float_as_uint(x);
  u += 0x7FFFu + ((u >> 16) & 1u);        // round-to-nearest-even
  return (unsigned short)(u >> 16);
}

// ---------------------------------------------------------------------------
// Generic strided f32 copy (h = x, z = h)
// ---------------------------------------------------------------------------
__global__ void k_copy_strided(const float* __restrict__ src, int ldsrc,
                               float* __restrict__ dst, int lddst,
                               long long rows, int cols) {
  long long i = (long long)blockIdx.x * blockDim.x + threadIdx.x;
  long long total = rows * cols;
  if (i >= total) return;
  long long r = i / cols;
  int c = (int)(i % cols);
  dst[r * lddst + c] = src[r * ldsrc + c];
}

// ---------------------------------------------------------------------------
// GIN aggregation: z[dst] += h[src]  (float4 gather + 4 f32 atomics per thread)
// ---------------------------------------------------------------------------
__global__ void k_scatter_add(const float* __restrict__ h,
                              const int* __restrict__ srcIdx,
                              const int* __restrict__ dstIdx,
                              float* __restrict__ z /* ld = 304 */) {
  long long i = (long long)blockIdx.x * blockDim.x + threadIdx.x;
  const long long total = (long long)N_EDGES * (DIM / 4);
  if (i >= total) return;
  int e = (int)(i / (DIM / 4));
  int chunk = (int)(i % (DIM / 4));
  int s = srcIdx[e], d = dstIdx[e];
  const float4 v = *(const float4*)(h + (long long)s * DIM + chunk * 4);
  float* zp = z + (long long)d * 304 + chunk * 4;
  atomicAdd(zp + 0, v.x);
  atomicAdd(zp + 1, v.y);
  atomicAdd(zp + 2, v.z);
  atomicAdd(zp + 3, v.w);
}

// ---------------------------------------------------------------------------
// f32 [rows, cols] (stride ldin) -> bf16 [rows, ldout] zero-padded K for WMMA
// ---------------------------------------------------------------------------
__global__ void k_cast_pad_bf16(const float* __restrict__ in, int ldin, int cols,
                                unsigned short* __restrict__ out, int ldout,
                                long long rows) {
  long long i = (long long)blockIdx.x * blockDim.x + threadIdx.x;
  long long total = rows * ldout;
  if (i >= total) return;
  long long r = i / ldout;
  int c = (int)(i % ldout);
  float v = (c < cols) ? in[r * ldin + c] : 0.0f;
  out[i] = f2bf(v);
}

// Zero the K-pad columns [c0, ld) of a bf16 activation buffer (done once per
// launch; the GEMM bf16 epilogue only writes the valid columns).
__global__ void k_zero_pad_cols_bf16(unsigned short* __restrict__ buf, int ld,
                                     int c0, long long rows) {
  const int w = ld - c0;
  long long i = (long long)blockIdx.x * blockDim.x + threadIdx.x;
  long long total = rows * w;
  if (i >= total) return;
  long long r = i / w;
  int c = c0 + (int)(i % w);
  buf[r * ld + c] = 0;
}

// ---------------------------------------------------------------------------
// Pack f32 weight [K, Ncols] into WMMA bf16 B-fragment layout:
//   frag (kf, nt) = 512 bf16, per-lane contiguous 16 values:
//   lane l holds column nt*16+l%16, K rows kf*32 + {8h..8h+7, 16+8h..16+8h+7}, h=l/16
// Zero-padded to kfrags*32 rows and ntilesP*16 cols.
// ---------------------------------------------------------------------------
__global__ void k_pack_weight(const float* __restrict__ W, int K, int Ncols,
                              unsigned short* __restrict__ out,
                              int kfrags, int ntilesP) {
  int i = blockIdx.x * blockDim.x + threadIdx.x;
  int total = kfrags * ntilesP * 512;
  if (i >= total) return;
  int kf = i / (ntilesP * 512);
  int r  = i % (ntilesP * 512);
  int nt = r / 512;
  int q  = r % 512;
  int lane = q / 16;           // lane within fragment
  int c    = q % 16;           // element within lane
  int k   = kf * 32 + (c < 8 ? 8 * (lane >> 4) + c : 16 + 8 * (lane >> 4) + (c - 8));
  int col = nt * 16 + (lane & 15);
  float v = (k < K && col < Ncols) ? W[(long long)k * Ncols + col] : 0.0f;
  out[i] = f2bf(v);
}

// ---------------------------------------------------------------------------
// TDM: stage a [rows x tileDwords] 2D panel of packed B-fragments into LDS.
// Descriptor per CDNA5 ISA ch.8 (D# groups). Dynamic LDS starts at offset 0
// (no static LDS in the GEMM kernel). data_size = 4B, OOB disabled via huge
// tensor dims. Issued by wave 0 only; waited with s_wait_tensorcnt.
// ---------------------------------------------------------------------------
__device__ __forceinline__ void tdm_load_panel(const unsigned short* gsrc,
                                               int tileDwords, int rows,
                                               int rowStrideDwords) {
  u32x4 g0;
  g0[0] = 1u;                                  // count=1 valid descriptor
  g0[1] = 0u;                                  // lds_addr = start of dynamic LDS
  unsigned long long ga = (unsigned long long)gsrc;
  g0[2] = (unsigned)(ga & 0xFFFFFFFFull);
  g0[3] = (unsigned)((ga >> 32) & 0x1FFFFFFull) | (2u << 30);  // type=2 (image)
  i32x8 g1;
  g1[0] = (int)(2u << 16);                     // data_size = 4 bytes; mask=0
  g1[1] = (int)0xFFFF0000u;                    // tensor_dim0 = 0x7FFFFFFF (lo16<<16)
  g1[2] = (int)0xFFFF7FFFu;                    // tensor_dim0 hi16 | tensor_dim1 lo16
  g1[3] = (int)(((unsigned)tileDwords << 16) | 0x7FFFu); // tensor_dim1 hi | tile_dim0
  g1[4] = rows;                                // tile_dim1 (K-fragment rows)
  g1[5] = rowStrideDwords;                     // tensor_dim0_stride (dwords)
  g1[6] = 0;
  g1[7] = 0;
  i32x4 gz = {0, 0, 0, 0};
#if defined(__clang_major__) && (__clang_major__ >= 23)
  i32x8 gz8 = {0, 0, 0, 0, 0, 0, 0, 0};
  __builtin_amdgcn_tensor_load_to_lds(g0, g1, gz, gz, gz8, 0);
#else
  __builtin_amdgcn_tensor_load_to_lds(g0, g1, gz, gz, 0);
#endif
  __builtin_amdgcn_s_wait_tensorcnt(0);
}

// ---------------------------------------------------------------------------
// bf16 WMMA GEMM:  C = act(A[M, lda](bf16) @ Bpack + bias)
//   - block = 8 waves; wave w owns M-tile (blockIdx.x*8 + w)
//   - block's B panel (8 N-tiles x all K frags) TDM-staged into LDS once
//   - per K-fragment: one A-fragment fetch, then two half-groups of
//     (4 B-fragment ds loads -> 4 WMMAs); sched_group_barrier describes the
//     desired issue pattern WITHOUT hard-fencing (a full fence made the
//     register allocator spill accumulators to scratch)
//   - epilogue writes f32 (Cf) or rounds straight to bf16 (Cb) so the next
//     GEMM's A operand never round-trips HBM in f32
// ---------------------------------------------------------------------------
__global__ void k_gemm_bf16(const unsigned short* __restrict__ A, int lda,
                            const unsigned short* __restrict__ Bpack,
                            int ntilesP, int kfrags, int mTiles,
                            const float* __restrict__ bias, int ncolsValid,
                            float* __restrict__ Cf, unsigned short* __restrict__ Cb,
                            int ldc, int doRelu, int outBf16) {
  extern __shared__ __align__(16) unsigned char smem[];
  const int wid  = threadIdx.x >> 5;
  const int lane = threadIdx.x & 31;

  if (threadIdx.x < 32u) {
    const unsigned short* src = Bpack + (long long)blockIdx.y * 8 * 512;
    tdm_load_panel(src, 8 * 256 /* 8 frags * 256 dwords */, kfrags,
                   ntilesP * 256 /* row stride in dwords */);
  }
  __syncthreads();

  const int mTile = blockIdx.x * 8 + wid;
  if (mTile >= mTiles) return;           // whole-wave early exit (EXEC stays full)

  const int half = lane >> 4;            // 0: K {0..7,16..23}; 1: K {8..15,24..31}
  const unsigned short* aRow = A + (long long)(mTile * 16 + (lane & 15)) * lda;

  v8f acc[8];
#pragma unroll
  for (int nt = 0; nt < 8; ++nt)
    acc[nt] = (v8f){0.f, 0.f, 0.f, 0.f, 0.f, 0.f, 0.f, 0.f};

#pragma unroll 1
  for (int kf = 0; kf < kfrags; ++kf) {
    const int k0 = kf * 32;
    v8bf alo = *(const v8bf*)(aRow + k0 + half * 8);
    v8bf ahi = *(const v8bf*)(aRow + k0 + 16 + half * 8);
    v16bf a = __builtin_shufflevector(alo, ahi, 0, 1, 2, 3, 4, 5, 6, 7,
                                      8, 9, 10, 11, 12, 13, 14, 15);
    const unsigned char* bbase = smem + (long long)kf * 8192 + lane * 32;

    // Half-group 0: fragments 0..3
    v16bf b0 = *(const v16bf*)(bbase + 0 * 1024);
    v16bf b1 = *(const v16bf*)(bbase + 1 * 1024);
    v16bf b2 = *(const v16bf*)(bbase + 2 * 1024);
    v16bf b3 = *(const v16bf*)(bbase + 3 * 1024);
    acc[0] = __builtin_amdgcn_wmma_f32_16x16x32_bf16(false, a, false, b0,
                                                     (short)0, acc[0], false, false);
    acc[1] = __builtin_amdgcn_wmma_f32_16x16x32_bf16(false, a, false, b1,
                                                     (short)0, acc[1], false, false);
    acc[2] = __builtin_amdgcn_wmma_f32_16x16x32_bf16(false, a, false, b2,
                                                     (short)0, acc[2], false, false);
    acc[3] = __builtin_amdgcn_wmma_f32_16x16x32_bf16(false, a, false, b3,
                                                     (short)0, acc[3], false, false);
    // Half-group 1: fragments 4..7
    v16bf b4 = *(const v16bf*)(bbase + 4 * 1024);
    v16bf b5 = *(const v16bf*)(bbase + 5 * 1024);
    v16bf b6 = *(const v16bf*)(bbase + 6 * 1024);
    v16bf b7 = *(const v16bf*)(bbase + 7 * 1024);
    acc[4] = __builtin_amdgcn_wmma_f32_16x16x32_bf16(false, a, false, b4,
                                                     (short)0, acc[4], false, false);
    acc[5] = __builtin_amdgcn_wmma_f32_16x16x32_bf16(false, a, false, b5,
                                                     (short)0, acc[5], false, false);
    acc[6] = __builtin_amdgcn_wmma_f32_16x16x32_bf16(false, a, false, b6,
                                                     (short)0, acc[6], false, false);
    acc[7] = __builtin_amdgcn_wmma_f32_16x16x32_bf16(false, a, false, b7,
                                                     (short)0, acc[7], false, false);

    // Desired per-iteration issue pattern (soft scheduling guidance):
    //   2x VMEM-read (A) -> 8x DS-read -> 4x WMMA -> 8x DS-read -> 4x WMMA
    __builtin_amdgcn_sched_group_barrier(0x020, 2, 0);  // VMEM read
    __builtin_amdgcn_sched_group_barrier(0x100, 8, 0);  // DS read
    __builtin_amdgcn_sched_group_barrier(0x008, 4, 0);  // MFMA/WMMA
    __builtin_amdgcn_sched_group_barrier(0x100, 8, 0);  // DS read
    __builtin_amdgcn_sched_group_barrier(0x008, 4, 0);  // MFMA/WMMA
  }

  // C/D layout: VGPR j, lanes 0-15 -> M = j, lanes 16-31 -> M = 8+j; N = lane%16
  const int colBase = blockIdx.y * 128 + (lane & 15);
#pragma unroll
  for (int nt = 0; nt < 8; ++nt) {
    const int col = colBase + nt * 16;
    if (col < ncolsValid) {
      const float bv = bias[col];
#pragma unroll
      for (int j = 0; j < 8; ++j) {
        const int row = mTile * 16 + half * 8 + j;
        float v = acc[nt][j] + bv;
        if (doRelu) v = fmaxf(v, 0.0f);
        if (outBf16) Cb[(long long)row * ldc + col] = f2bf(v);
        else         Cf[(long long)row * ldc + col] = v;
      }
    }
  }
}

// ---------------------------------------------------------------------------
// BatchNorm stats, deterministic + coalesced, two stages:
//  stage 1: block = row chunk, thread = channel (coalesced across the wave)
//  stage 2: block = channel, tree-reduce the 125 chunk partials
// ---------------------------------------------------------------------------
__global__ void k_bn_partial(const float* __restrict__ z, int ldz,
                             float* __restrict__ psum, float* __restrict__ psq) {
  const int c = threadIdx.x;
  if (c >= DIM) return;
  const int r0 = blockIdx.x * BN_ROWS_PER_CHUNK;
  float sum = 0.f, sq = 0.f;
  for (int r = r0; r < r0 + BN_ROWS_PER_CHUNK; ++r) {
    float v = z[(long long)r * ldz + c];
    sum += v;
    sq += v * v;
  }
  psum[(long long)blockIdx.x * DIM + c] = sum;
  psq[(long long)blockIdx.x * DIM + c]  = sq;
}

__global__ void k_bn_finish(const float* __restrict__ psum,
                            const float* __restrict__ psq,
                            float* __restrict__ mu, float* __restrict__ rsig) {
  __shared__ float s1[128];
  __shared__ float s2[128];
  const int c = blockIdx.x;
  const int t = threadIdx.x;
  float sum = 0.f, sq = 0.f;
  if (t < BN_CHUNKS) {
    sum = psum[(long long)t * DIM + c];
    sq  = psq[(long long)t * DIM + c];
  }
  s1[t] = sum;
  s2[t] = sq;
  __syncthreads();
  for (int s = 64; s > 0; s >>= 1) {
    if (t < s) {
      s1[t] += s1[t + s];
      s2[t] += s2[t + s];
    }
    __syncthreads();
  }
  if (t == 0) {
    float m = s1[0] / (float)N_NODES;
    float var = s2[0] / (float)N_NODES - m * m;
    mu[c] = m;
    rsig[c] = rsqrtf(var + EPS_F);
  }
}

__global__ void k_bn_apply(const float* __restrict__ z, int ldz,
                           const float* __restrict__ mu,
                           const float* __restrict__ rsig,
                           const float* __restrict__ g,
                           const float* __restrict__ b,
                           float* __restrict__ h, int doRelu) {
  long long i = (long long)blockIdx.x * blockDim.x + threadIdx.x;
  const long long total = (long long)N_NODES * DIM;
  if (i >= total) return;
  long long r = i / DIM;
  int c = (int)(i % DIM);
  float v = (z[r * ldz + c] - mu[c]) * rsig[c] * g[c] + b[c];
  if (doRelu) v = fmaxf(v, 0.0f);
  h[i] = v;
}

// ---------------------------------------------------------------------------
// global_mean_pool: batch is sorted -> binary-search segment, deterministic sum
// (at each inner row step, lanes read consecutive channels -> coalesced)
// ---------------------------------------------------------------------------
__device__ __forceinline__ int lower_bound_i32(const int* a, int n, int key) {
  int lo = 0, hi = n;
  while (lo < hi) {
    int mid = (lo + hi) >> 1;
    if (a[mid] < key) lo = mid + 1; else hi = mid;
  }
  return lo;
}

__global__ void k_pool(const float* __restrict__ h, const int* __restrict__ batch,
                       float* __restrict__ hg) {
  const int b = blockIdx.x;
  const int lo = lower_bound_i32(batch, N_NODES, b);
  const int hi = lower_bound_i32(batch, N_NODES, b + 1);
  const float inv = 1.0f / fmaxf((float)(hi - lo), 1.0f);
  for (int c = threadIdx.x; c < DIM; c += blockDim.x) {
    float s = 0.f;
    for (int r = lo; r < hi; ++r) s += h[(long long)r * DIM + c];
    hg[(long long)b * DIM + c] = s * inv;
  }
}

// ---------------------------------------------------------------------------
// Final LayerNorm over NOUT=768, one block per row
// ---------------------------------------------------------------------------
__global__ void k_layernorm(const float* __restrict__ y,
                            const float* __restrict__ g,
                            const float* __restrict__ b,
                            float* __restrict__ out) {
  __shared__ float s1[256];
  __shared__ float s2[256];
  const int r = blockIdx.x;
  float sum = 0.f, sq = 0.f;
  for (int c = threadIdx.x; c < NOUT; c += blockDim.x) {
    float v = y[(long long)r * NOUT + c];
    sum += v;
    sq += v * v;
  }
  s1[threadIdx.x] = sum;
  s2[threadIdx.x] = sq;
  __syncthreads();
  for (int s = 128; s > 0; s >>= 1) {
    if ((int)threadIdx.x < s) {
      s1[threadIdx.x] += s1[threadIdx.x + s];
      s2[threadIdx.x] += s2[threadIdx.x + s];
    }
    __syncthreads();
  }
  __shared__ float mS, rsS;
  if (threadIdx.x == 0) {
    float m = s1[0] / (float)NOUT;
    mS = m;
    rsS = rsqrtf(s2[0] / (float)NOUT - m * m + EPS_F);
  }
  __syncthreads();
  const float m = mS, rs = rsS;
  for (int c = threadIdx.x; c < NOUT; c += blockDim.x) {
    out[(long long)r * NOUT + c] =
        (y[(long long)r * NOUT + c] - m) * rs * g[c] + b[c];
  }
}

// ---------------------------------------------------------------------------
// Host-side orchestration
// ---------------------------------------------------------------------------
static inline unsigned nblk(long long total, int tb) {
  return (unsigned)((total + tb - 1) / tb);
}

extern "C" void kernel_launch(void* const* d_in, const int* in_sizes, int n_in,
                              void* d_out, int out_size, void* d_ws, size_t ws_size,
                              hipStream_t stream) {
  (void)in_sizes; (void)n_in; (void)out_size; (void)ws_size;
  const float* x    = (const float*)d_in[0];
  const int*   ei   = (const int*)d_in[1];
  const int*   bat  = (const int*)d_in[2];
  const float* W1s  = (const float*)d_in[3];
  const float* b1s  = (const float*)d_in[4];
  const float* W2s  = (const float*)d_in[5];
  const float* b2s  = (const float*)d_in[6];
  const float* bn_g = (const float*)d_in[7];
  const float* bn_b = (const float*)d_in[8];
  const float* P1w  = (const float*)d_in[9];
  const float* P1b  = (const float*)d_in[10];
  const float* P2w  = (const float*)d_in[11];
  const float* P2b  = (const float*)d_in[12];
  const float* ln_g = (const float*)d_in[13];
  const float* ln_b = (const float*)d_in[14];
  const int* srcIdx = ei;
  const int* dstIdx = ei + N_EDGES;

  // Workspace carve-out (~230 MB)
  char* ws = (char*)d_ws;
  size_t off = 0;
  auto carve = [&](size_t bytes) -> void* {
    void* p = ws + off;
    off = (off + bytes + 255) & ~(size_t)255;
    return p;
  };
  float* h      = (float*)carve((size_t)N_NODES * 300 * 4);
  float* z      = (float*)carve((size_t)N_NODES * 304 * 4);   // agg / gemm2 out
  unsigned short* abf  = (unsigned short*)carve((size_t)N_NODES * 320 * 2);
  unsigned short* mbf  = (unsigned short*)carve((size_t)N_NODES * 640 * 2);
  unsigned short* w1p  = (unsigned short*)carve((size_t)10 * 40 * 512 * 2);
  unsigned short* w2p  = (unsigned short*)carve((size_t)20 * 24 * 512 * 2);
  unsigned short* p1p  = (unsigned short*)carve((size_t)10 * 32 * 512 * 2);
  unsigned short* p2p  = (unsigned short*)carve((size_t)16 * 48 * 512 * 2);
  float* psum = (float*)carve((size_t)BN_CHUNKS * DIM * 4);
  float* psq  = (float*)carve((size_t)BN_CHUNKS * DIM * 4);
  float* mu   = (float*)carve(304 * 4);
  float* rsig = (float*)carve(304 * 4);
  float* hg   = (float*)carve((size_t)NB * 300 * 4);
  unsigned short* hgbf = (unsigned short*)carve((size_t)NB * 320 * 2);
  unsigned short* y1bf = (unsigned short*)carve((size_t)NB * 512 * 2);
  float* ybuf = (float*)carve((size_t)NB * 768 * 4);

  const int TB = 256;
  const int MT_NODES = N_NODES / 16;   // 3125 M-tiles
  const int MT_B     = NB / 16;        // 128 M-tiles

  // h = x ; zero K-pad columns of mbf once (GEMM1 bf16 epilogue skips them)
  k_copy_strided<<<nblk((long long)N_NODES * 300, TB), TB, 0, stream>>>(
      x, 300, h, 300, N_NODES, 300);
  k_zero_pad_cols_bf16<<<nblk((long long)N_NODES * 40, TB), TB, 0, stream>>>(
      mbf, 640, 600, N_NODES);
  // Projection weights packed once
  k_pack_weight<<<nblk(10 * 32 * 512, TB), TB, 0, stream>>>(P1w, 300, 512, p1p, 10, 32);
  k_pack_weight<<<nblk(16 * 48 * 512, TB), TB, 0, stream>>>(P2w, 512, 768, p2p, 16, 48);

  for (int i = 0; i < NLAYERS; ++i) {
    const float* W1 = W1s + (size_t)i * 300 * 600;
    const float* B1 = b1s + (size_t)i * 600;
    const float* W2 = W2s + (size_t)i * 600 * 300;
    const float* B2 = b2s + (size_t)i * 300;
    const float* G  = bn_g + (size_t)i * 300;
    const float* Bb = bn_b + (size_t)i * 300;

    k_pack_weight<<<nblk(10 * 40 * 512, TB), TB, 0, stream>>>(W1, 300, 600, w1p, 10, 40);
    k_pack_weight<<<nblk(20 * 24 * 512, TB), TB, 0, stream>>>(W2, 600, 300, w2p, 20, 24);

    // z = h + scatter_add(h[src] -> dst)
    k_copy_strided<<<nblk((long long)N_NODES * 300, TB), TB, 0, stream>>>(
        h, 300, z, 304, N_NODES, 300);
    k_scatter_add<<<nblk((long long)N_EDGES * (DIM / 4), TB), TB, 0, stream>>>(
        h, srcIdx, dstIdx, z);

    // GEMM1: mbf = bf16(relu(z @ W1 + b1))   M=50000 K=320 N=640(pad,600 valid)
    k_cast_pad_bf16<<<nblk((long long)N_NODES * 320, TB), TB, 0, stream>>>(
        z, 304, 300, abf, 320, N_NODES);
    {
      dim3 g((MT_NODES + 7) / 8, 5);
      k_gemm_bf16<<<g, 256, 10 * 8192, stream>>>(abf, 320, w1p, 40, 10, MT_NODES,
                                                 B1, 600, nullptr, mbf, 640, 1, 1);
    }

    // GEMM2: z = mbf @ W2 + b2 (f32 for BN)  M=50000 K=640 N=304(pad,300 valid)
    {
      dim3 g((MT_NODES + 7) / 8, 3);
      k_gemm_bf16<<<g, 256, 20 * 8192, stream>>>(mbf, 640, w2p, 24, 20, MT_NODES,
                                                 B2, 300, z, nullptr, 304, 0, 0);
    }

    // BatchNorm (+ ReLU except last layer) -> h
    k_bn_partial<<<BN_CHUNKS, 320, 0, stream>>>(z, 304, psum, psq);
    k_bn_finish<<<DIM, 128, 0, stream>>>(psum, psq, mu, rsig);
    k_bn_apply<<<nblk((long long)N_NODES * DIM, TB), TB, 0, stream>>>(
        z, 304, mu, rsig, G, Bb, h, (i < NLAYERS - 1) ? 1 : 0);
  }

  // Pool + projection head + LayerNorm
  k_pool<<<NB, 256, 0, stream>>>(h, bat, hg);
  k_cast_pad_bf16<<<nblk((long long)NB * 320, TB), TB, 0, stream>>>(
      hg, 300, 300, hgbf, 320, NB);
  {
    dim3 g(MT_B / 8, 4);   // M=2048 K=320 N=512 -> bf16 out (all cols valid)
    k_gemm_bf16<<<g, 256, 10 * 8192, stream>>>(hgbf, 320, p1p, 32, 10, MT_B,
                                               P1b, 512, nullptr, y1bf, 512, 1, 1);
  }
  {
    dim3 g(MT_B / 8, 6);   // M=2048 K=512 N=768 -> f32 out
    k_gemm_bf16<<<g, 256, 16 * 8192, stream>>>(y1bf, 512, p2p, 48, 16, MT_B,
                                               P2b, 768, ybuf, nullptr, 768, 0, 0);
  }
  k_layernorm<<<NB, 256, 0, stream>>>(ybuf, ln_g, ln_b, (float*)d_out);
}
